// AdjMtxReconstructor_24386824306744
// MI455X (gfx1250) — compile-verified, matrix operand order checked by reading, not run on hardware
//
#include <hip/hip_runtime.h>
#include <hip/hip_bf16.h>
#include <math.h>

typedef __attribute__((ext_vector_type(16))) _Float16 v16h;
typedef __attribute__((ext_vector_type(8)))  _Float16 v8h;
typedef __attribute__((ext_vector_type(8)))  float    v8f;

#define NUM_CARDS 50000
#define EMBED_DIM 64
#define HIDDEN    256
#define BATCH     1024
#define NM1       (NUM_CARDS - 1)            /* 49999 valid cards: indices 1..49999 */
#define CARD_TILES ((NM1 + 15) / 16)         /* 3125 tiles of 16 cards */
#define KL_EPS      1e-7f
#define LOG2_KL_EPS (-23.253496664211536f)   /* log2(1e-7) */
#define LOG2E       1.4426950408889634f
#define GRID_Y      32                       /* card-dimension block split  */
#define WPB         8                        /* waves per block (256 thr)   */
#define TILE_STRIDE (GRID_Y * WPB)           /* 256: compile-time constant  */

// ---------------------------------------------------------------------------
// Kernel 1: gather + MLP + L2-normalize -> a_f16 (1024 x 64, half)
// ---------------------------------------------------------------------------
__global__ void prep_a_kernel(const int* __restrict__ single_card,
                              const float* __restrict__ CE,
                              const float* __restrict__ W1,
                              const float* __restrict__ b1,
                              const float* __restrict__ W2,
                              const float* __restrict__ b2,
                              _Float16* __restrict__ a_h) {
  __shared__ float emb[EMBED_DIM];
  __shared__ float h[HIDDEN];
  __shared__ float o[EMBED_DIM];
  __shared__ float scale_s;
  const int row = blockIdx.x;
  const int t = threadIdx.x;
  const int idx = single_card[row];
  if (t < EMBED_DIM) emb[t] = CE[idx * EMBED_DIM + t];
  __syncthreads();
  float acc = b1[t];
#pragma unroll 8
  for (int k = 0; k < EMBED_DIM; ++k) acc += emb[k] * W1[k * HIDDEN + t];
  h[t] = fmaxf(acc, 0.0f);
  __syncthreads();
  if (t < EMBED_DIM) {
    float a2 = b2[t];
#pragma unroll 8
    for (int k = 0; k < HIDDEN; ++k) a2 += h[k] * W2[k * EMBED_DIM + t];
    o[t] = a2;
  }
  __syncthreads();
  if (t == 0) {
    float s = 0.f;
    for (int k = 0; k < EMBED_DIM; ++k) s += o[k] * o[k];
    scale_s = rsqrtf(fmaxf(s, 1e-12f));
  }
  __syncthreads();
  if (t < EMBED_DIM) a_h[row * EMBED_DIM + t] = (_Float16)(o[t] * scale_s);
}

// ---------------------------------------------------------------------------
// Kernel 2: L2-normalize card embeddings -> b_f16 (50001 x 64 half; row 50000
// is a zero pad row so the partial last WMMA tile yields finite c=0).
// ---------------------------------------------------------------------------
__global__ void prep_b_kernel(const float* __restrict__ CE,
                              _Float16* __restrict__ b_h) {
  const int wave = threadIdx.x >> 5;
  const int lane = threadIdx.x & 31;
  const int row = blockIdx.x * 8 + wave;
  if (row > NUM_CARDS) return;
  if (row == NUM_CARDS) {                       // zero pad row
    b_h[row * 64 + lane] = (_Float16)0.f;
    b_h[row * 64 + lane + 32] = (_Float16)0.f;
    return;
  }
  const float x0 = CE[row * 64 + lane];
  const float x1 = CE[row * 64 + lane + 32];
  float ss = x0 * x0 + x1 * x1;
  for (int off = 16; off; off >>= 1) ss += __shfl_xor(ss, off, 32);
  const float sc = rsqrtf(fmaxf(ss, 1e-12f));
  b_h[row * 64 + lane]      = (_Float16)(x0 * sc);
  b_h[row * 64 + lane + 32] = (_Float16)(x1 * sc);
}

// ---------------------------------------------------------------------------
__global__ void zero_kernel(float* __restrict__ Z, float* __restrict__ accum) {
  const int t = blockIdx.x * blockDim.x + threadIdx.x;
  if (t < BATCH) Z[t] = 0.f;
  if (t == 0) accum[0] = 0.f;
}

// ---------------------------------------------------------------------------
// WMMA 16x32 f16 fragment per ISA 7.12.2: lane l -> row/col l&15,
// kb = 8*(l>>4); elems 0..7 = K kb..kb+7, elems 8..15 = K kb+16..kb+23.
// ---------------------------------------------------------------------------
__device__ inline v16h load_frag(const _Float16* __restrict__ base, int kb) {
  union { v16h v; v8h h[2]; } u;
  u.h[0] = *(const v8h*)(base + kb);
  u.h[1] = *(const v8h*)(base + kb + 16);
  return u.v;
}

// ---------------------------------------------------------------------------
// Phase 1: Z[i] = sum_j 2^(ts2*cos(i,j)), ts2 = t*log2(e).  exp args bounded
// (|cos|<=1, t=5) so no max pass.  Double-buffered B fragments: next tile's
// loads are issued before the current WMMAs so s_wait overlaps compute.
// ---------------------------------------------------------------------------
__global__ void __launch_bounds__(256)
phase1_kernel(const _Float16* __restrict__ a_h,
              const _Float16* __restrict__ b_h,
              const float* __restrict__ temp,
              float* __restrict__ Z) {
  const float ts2 = temp[0] * LOG2E;
  const int lane = threadIdx.x & 31;
  const int wave = threadIdx.x >> 5;
  const int rt = blockIdx.x;
  const int m = lane & 15;
  const int hi = lane >> 4;
  const int kb = hi * 8;
  const _Float16* aptr = a_h + (rt * 16 + m) * 64;
  const v16h A0 = load_frag(aptr, kb);
  const v16h A1 = load_frag(aptr, kb + 32);
  float zs[8];
#pragma unroll
  for (int r = 0; r < 8; ++r) zs[r] = 0.f;

  const int gw = blockIdx.y * WPB + wave;
  const size_t binc = (size_t)TILE_STRIDE * 16 * 64;   // halves per tile-stride
  const _Float16* bptr = b_h + (size_t)(1 + gw * 16 + m) * 64;
  v16h B0 = load_frag(bptr, kb);
  v16h B1 = load_frag(bptr, kb + 32);

  for (int tile = gw; tile < CARD_TILES; tile += TILE_STRIDE) {
    const bool more = (tile + TILE_STRIDE) < CARD_TILES;     // wave-uniform
    const _Float16* nb = more ? (bptr + binc) : bptr;
    const v16h N0 = load_frag(nb, kb);
    const v16h N1 = load_frag(nb, kb + 32);

    v8f c = {};
    c = __builtin_amdgcn_wmma_f32_16x16x32_f16(false, A0, false, B0, (short)0, c, false, false);
    c = __builtin_amdgcn_wmma_f32_16x16x32_f16(false, A1, false, B1, (short)0, c, false, false);

    const float vm = ((1 + tile * 16 + m) <= NM1) ? 1.0f : 0.0f;  // no EXEC churn
#pragma unroll
    for (int r = 0; r < 8; ++r)
      zs[r] += __builtin_amdgcn_exp2f(ts2 * c[r]) * vm;

    B0 = N0; B1 = N1; bptr = nb;
  }
#pragma unroll
  for (int r = 0; r < 8; ++r) {
    float v = zs[r];
    v += __shfl_xor(v, 1, 32);
    v += __shfl_xor(v, 2, 32);
    v += __shfl_xor(v, 4, 32);
    v += __shfl_xor(v, 8, 32);
    zs[r] = v;
  }
  if (m == 0) {
    const int rbase = rt * 16 + hi * 8;
#pragma unroll
    for (int r = 0; r < 8; ++r) atomicAdd(&Z[rbase + r], zs[r]);
  }
}

// ---------------------------------------------------------------------------
// Phase 2: recompute similarities, fused clipped KL in log2 space (sum is
// directly in bits): contrib = y * (log2 y - clamp(ts2*s - log2 Z, log2 eps, 0))
// Streams adj_row (205 MB) exactly once.
// ---------------------------------------------------------------------------
__global__ void __launch_bounds__(256)
phase2_kernel(const _Float16* __restrict__ a_h,
              const _Float16* __restrict__ b_h,
              const float* __restrict__ adj,
              const float* __restrict__ temp,
              const float* __restrict__ Z,
              float* __restrict__ accum) {
  const float ts2 = temp[0] * LOG2E;
  const int lane = threadIdx.x & 31;
  const int wave = threadIdx.x >> 5;
  const int rt = blockIdx.x;
  const int m = lane & 15;
  const int hi = lane >> 4;
  const int kb = hi * 8;
  const _Float16* aptr = a_h + (rt * 16 + m) * 64;
  const v16h A0 = load_frag(aptr, kb);
  const v16h A1 = load_frag(aptr, kb + 32);
  float lz[8];
#pragma unroll
  for (int r = 0; r < 8; ++r)
    lz[r] = __builtin_amdgcn_logf(Z[rt * 16 + hi * 8 + r]);   // log2(Z)
  float kl = 0.f;

  const int gw = blockIdx.y * WPB + wave;
  const size_t binc = (size_t)TILE_STRIDE * 16 * 64;
  const _Float16* bptr = b_h + (size_t)(1 + gw * 16 + m) * 64;
  v16h B0 = load_frag(bptr, kb);
  v16h B1 = load_frag(bptr, kb + 32);

  for (int tile = gw; tile < CARD_TILES; tile += TILE_STRIDE) {
    const bool more = (tile + TILE_STRIDE) < CARD_TILES;
    const _Float16* nb = more ? (bptr + binc) : bptr;
    const v16h N0 = load_frag(nb, kb);
    const v16h N1 = load_frag(nb, kb + 32);

    v8f c = {};
    c = __builtin_amdgcn_wmma_f32_16x16x32_f16(false, A0, false, B0, (short)0, c, false, false);
    c = __builtin_amdgcn_wmma_f32_16x16x32_f16(false, A1, false, B1, (short)0, c, false, false);

    const int card = 1 + tile * 16 + m;
    const float vm = (card <= NM1) ? 1.0f : 0.0f;
    const int col = (card <= NM1 ? card : NM1) - 1;          // clamped adj col
#pragma unroll
    for (int r = 0; r < 8; ++r) {
      const int row = rt * 16 + hi * 8 + r;
      const float y = adj[(size_t)row * NM1 + col];
      const float yc = fminf(fmaxf(y, KL_EPS), 1.0f);
      const float lp = fminf(fmaxf(ts2 * c[r] - lz[r], LOG2_KL_EPS), 0.0f);
      kl += vm * yc * (__builtin_amdgcn_logf(yc) - lp);
    }
    B0 = N0; B1 = N1; bptr = nb;
  }
  for (int off = 16; off; off >>= 1) kl += __shfl_xor(kl, off, 32);
  if (lane == 0) atomicAdd(accum, kl);
}

// ---------------------------------------------------------------------------
// Finalize: accum is already in bits -> loss = accum/BATCH + t^2 * 0.01
// ---------------------------------------------------------------------------
__global__ void finalize_kernel(const float* __restrict__ accum,
                                const float* __restrict__ temp,
                                float* __restrict__ out) {
  const float t = temp[0];
  out[0] = accum[0] * (1.0f / BATCH) + t * t * 0.01f;
}

// ---------------------------------------------------------------------------
extern "C" void kernel_launch(void* const* d_in, const int* in_sizes, int n_in,
                              void* d_out, int out_size, void* d_ws, size_t ws_size,
                              hipStream_t stream) {
  (void)in_sizes; (void)n_in; (void)out_size; (void)ws_size;
  const int*   single_card = (const int*)  d_in[0];
  const float* adj_row     = (const float*)d_in[1];
  const float* card_emb    = (const float*)d_in[2];
  const float* W1          = (const float*)d_in[3];
  const float* b1          = (const float*)d_in[4];
  const float* W2          = (const float*)d_in[5];
  const float* b2          = (const float*)d_in[6];
  const float* temperature = (const float*)d_in[7];
  float* out = (float*)d_out;

  // workspace layout (16B aligned): a_f16 | b_f16 (+1 pad row) | Z | accum
  char* ws = (char*)d_ws;
  _Float16* a_h   = (_Float16*)(ws);                        // 1024*64*2        = 131072 B
  _Float16* b_h   = (_Float16*)(ws + 131072);               // 50001*64*2       = 6400128 B
  float*    Zbuf  = (float*)   (ws + 131072 + 6400128);     // 1024*4           = 4096 B
  float*    accum = (float*)   (ws + 131072 + 6400128 + 4096);

  zero_kernel<<<(BATCH + 255) / 256, 256, 0, stream>>>(Zbuf, accum);
  prep_a_kernel<<<BATCH, HIDDEN, 0, stream>>>(single_card, card_emb, W1, b1, W2, b2, a_h);
  prep_b_kernel<<<(NUM_CARDS + 1 + 7) / 8, 256, 0, stream>>>(card_emb, b_h);

  dim3 grid(BATCH / 16, GRID_Y);
  phase1_kernel<<<grid, 256, 0, stream>>>(a_h, b_h, temperature, Zbuf);
  phase2_kernel<<<grid, 256, 0, stream>>>(a_h, b_h, adj_row, temperature, Zbuf, accum);
  finalize_kernel<<<1, 1, 0, stream>>>(accum, temperature, out);
}
